// HybridPhotonicSystem_85237920956467
// MI455X (gfx1250) — matrix operand order repaired and checked by reading, not compiled, not run on hardware
//
#include <hip/hip_runtime.h>

// ---- types for WMMA ----
typedef __bf16  bf16;
typedef __bf16  v8bf  __attribute__((ext_vector_type(8)));
typedef __bf16  v16bf __attribute__((ext_vector_type(16)));
typedef float   v8f   __attribute__((ext_vector_type(8)));

// ---- problem constants (match reference) ----
#define N_IN   512
#define N_OUT  512
#define N_WL   64
#define BATCH  256

#define WL_MIN      1.53e-06f
#define WL_MAX      1.57e-06f
#define TWO_PI      6.2831853071795864769f
#define RING_RADIUS 5e-06f
#define N_GROUP     4.2f
#define ALPHA_LOSS  0.99979061035f   /* 1 - pi/15000 */
#define KERR_COEFF  2.7e-18f

// ---- tiling ----
#define TM 128      // batch tile (M)
#define TN 128      // n_out tile (N)
#define KC 32       // k chunk == WMMA K
#define KS 40       // padded LDS row stride in bf16 (80B = 5*16B, conflict-free)

// ======================================================================
// Kernel 1: kerr_phase[j] = KERR_COEFF * mean_{b,w}(input[b,j,w]^2) * 0.1
// ======================================================================
__global__ __launch_bounds__(256) void kerr_kernel(const float* __restrict__ in,
                                                   float* __restrict__ kerr) {
    const int j = blockIdx.x;
    float s = 0.0f;
    // BATCH*N_WL = 16384 elements for this column j; coalesced over w
    for (int f = threadIdx.x; f < BATCH * N_WL; f += 256) {
        const int b = f >> 6;        // / N_WL
        const int w = f & (N_WL - 1);
        const float v = in[(size_t)b * (N_IN * N_WL) + (size_t)j * N_WL + w];
        s += v * v;
    }
    __shared__ float red[256];
    red[threadIdx.x] = s;
    __syncthreads();
    for (int off = 128; off > 0; off >>= 1) {
        if (threadIdx.x < off) red[threadIdx.x] += red[threadIdx.x + off];
        __syncthreads();
    }
    if (threadIdx.x == 0)
        kerr[j] = KERR_COEFF * (red[0] * (1.0f / (float)(BATCH * N_WL))) * 0.1f;
}

// ======================================================================
// Kernel 2: fused drop-transmission + per-wavelength GEMM via WMMA bf16
//   out[b,i,w] = sum_j in[b,j,w] * drop[i,j,w]
// grid = (N_WL, N_OUT/TN, BATCH/TM), block = 256 (8 wave32)
// ======================================================================
__global__ __launch_bounds__(256) void mrr_wmma_kernel(
    const float* __restrict__ in,     // [BATCH, N_IN, N_WL]
    const float* __restrict__ phase,  // [N_OUT, N_IN]
    const float* __restrict__ coup,   // [N_OUT, N_IN]
    const float* __restrict__ kerr,   // [N_IN]
    float* __restrict__ out)          // [BATCH, N_OUT, N_WL]
{
    const int w     = blockIdx.x;
    const int iTile = blockIdx.y * TN;
    const int bTile = blockIdx.z * TM;
    const int tid   = threadIdx.x;
    const int lane  = tid & 31;
    const int wave  = tid >> 5;
    const int waveM = (wave & 1) * 64;   // 2 waves along M
    const int waveN = (wave >> 1) * 32;  // 4 waves along N

    __shared__ bf16 As[TM][KS];  // A tile: [m][k] (bf16 input)
    __shared__ bf16 Bs[TN][KS];  // B tile: [n][k] (bf16 drop, k contiguous)

    const float wl = WL_MIN + (WL_MAX - WL_MIN) * (float)w / (float)(N_WL - 1);
    const float fsr_scale = 1.0f / (N_GROUP * TWO_PI * RING_RADIUS);

    v8f acc[4][2] = {};  // per-wave 64x32 C tile as 4x2 WMMA fragments

    const int ldR = tid >> 5;   // 0..7  (row group per staging pass)
    const int ldK = tid & 31;   // 0..31 (k within chunk)

    for (int j0 = 0; j0 < N_IN; j0 += KC) {
        const int j = j0 + ldK;
        // ---- stage A: gather input[bTile..+128][j0..+32] at fixed w ----
#pragma unroll
        for (int r = 0; r < 16; ++r) {
            const int m = r * 8 + ldR;
            const float v = in[((size_t)(bTile + m) * N_IN + j) * N_WL + w];
            As[m][ldK] = (bf16)v;
        }
        // ---- stage B: compute drop[iTile+n][j][w] ----
        const float kerr_j = kerr[j];
#pragma unroll
        for (int r = 0; r < 16; ++r) {
            const int n  = r * 8 + ldR;
            const int i  = iTile + n;
            const int ij = i * N_IN + j;
            const float cw  = WL_MIN + (WL_MAX - WL_MIN) * (float)ij *
                              (1.0f / (float)(N_IN * N_OUT));
            const float fsr = cw * cw * fsr_scale;
            const float det = TWO_PI * (wl - cw) / fsr;
            const float tp  = det + phase[ij] + kerr_j;
            const float kap = fminf(fmaxf(coup[ij], 0.1f), 0.9f);
            const float t   = sqrtf(1.0f - kap * kap);
            const float at  = ALPHA_LOSS * t;
            const float c   = __cosf(tp);
            const float sn  = __sinf(tp);
            const float d1  = 1.0f - at * c;
            const float den = d1 * d1 + at * at * sn * sn;
            Bs[n][ldK] = (bf16)(kap * kap * ALPHA_LOSS / den);
        }
        __syncthreads();

        // ---- fragments (ISA 16-bit A 16x32 / B 32x16 layouts, wave32) ----
        const int rowA  = lane & 15;
        const int halfA = (lane >> 4) * 8;   // lanes 0-15: K{0-7,16-23}; 16-31: K{8-15,24-31}
        v16bf af[4];
#pragma unroll
        for (int a = 0; a < 4; ++a) {
            const int m = waveM + 16 * a + rowA;
            const v8bf lo = *(const v8bf*)&As[m][halfA];
            const v8bf hi = *(const v8bf*)&As[m][halfA + 16];
            af[a] = __builtin_shufflevector(lo, hi, 0, 1, 2, 3, 4, 5, 6, 7,
                                            8, 9, 10, 11, 12, 13, 14, 15);
        }
        const int kB = (lane >> 4) * 8;      // lanes 0-15: K{0-7,16-23}; 16-31: K{8-15,24-31}
        v16bf bfr[2];
#pragma unroll
        for (int bb = 0; bb < 2; ++bb) {
            const int n = waveN + 16 * bb + rowA;
            const v8bf lo = *(const v8bf*)&Bs[n][kB];
            const v8bf hi = *(const v8bf*)&Bs[n][kB + 16];
            bfr[bb] = __builtin_shufflevector(lo, hi, 0, 1, 2, 3, 4, 5, 6, 7,
                                              8, 9, 10, 11, 12, 13, 14, 15);
        }
        // ---- 8 WMMAs per wave per k-chunk ----
#pragma unroll
        for (int a = 0; a < 4; ++a)
#pragma unroll
            for (int bb = 0; bb < 2; ++bb)
                acc[a][bb] = __builtin_amdgcn_wmma_f32_16x16x32_bf16(
                    false, af[a], false, bfr[bb], (short)0, acc[a][bb],
                    false, false);

        __syncthreads();
    }

    // ---- epilogue: C layout VGPR r -> rows {r, r+8}, col = lane%16 ----
    const int col    = lane & 15;
    const int rowOff = (lane >> 4) * 8;
#pragma unroll
    for (int a = 0; a < 4; ++a) {
#pragma unroll
        for (int bb = 0; bb < 2; ++bb) {
            const int i = iTile + waveN + 16 * bb + col;
#pragma unroll
            for (int r = 0; r < 8; ++r) {
                const int b = bTile + waveM + 16 * a + rowOff + r;
                out[((size_t)b * N_OUT + i) * N_WL + w] = acc[a][bb][r];
            }
        }
    }
}

// ======================================================================
extern "C" void kernel_launch(void* const* d_in, const int* in_sizes, int n_in,
                              void* d_out, int out_size, void* d_ws, size_t ws_size,
                              hipStream_t stream) {
    const float* in    = (const float*)d_in[0];  // [256,512,64] f32
    const float* phase = (const float*)d_in[1];  // [512,512] f32
    const float* coup  = (const float*)d_in[2];  // [512,512] f32
    float* kerr = (float*)d_ws;                  // 512 floats scratch
    float* out  = (float*)d_out;                 // [256,512,64] f32

    kerr_kernel<<<N_IN, 256, 0, stream>>>(in, kerr);

    dim3 grid(N_WL, N_OUT / TN, BATCH / TM);     // (64, 4, 2)
    mrr_wmma_kernel<<<grid, 256, 0, stream>>>(in, phase, coup, kerr, out);
}
